// FineI2P_61323543052708
// MI455X (gfx1250) — compile-verified, matrix operand order checked by reading, not run on hardware
//
#include <hip/hip_runtime.h>
#include <hip/hip_bf16.h>

// ---------------- problem constants ----------------
#define BB   2
#define NN   2048
#define MM   2048
#define DD   512
#define HH   8
#define DH   64
#define FF   2048
#define ROWS (BB * NN)   // 4096 (also BB*MM)

typedef __attribute__((ext_vector_type(16))) __bf16 v16bf;
typedef __attribute__((ext_vector_type(8)))  __bf16 bf16x8;
typedef __attribute__((ext_vector_type(8)))  float  v8f;
typedef __bf16 bf16;

// Load one A/B WMMA fragment (16x32 bf16, per-lane view):
// elements 0..7  = p[0..7]   (K = kb .. kb+7)
// elements 8..15 = p[16..23] (K = kb+16 .. kb+23)
__device__ __forceinline__ v16bf frag_ld(const bf16* p) {
    bf16x8 lo = *(const bf16x8*)(p);
    bf16x8 hi = *(const bf16x8*)(p + 16);
    v16bf f;
#pragma unroll
    for (int i = 0; i < 8; ++i) { f[i] = lo[i]; f[i + 8] = hi[i]; }
    return f;
}

__device__ __forceinline__ v8f wmma_bf16(v16bf a, v16bf b, v8f c) {
    return __builtin_amdgcn_wmma_f32_16x16x32_bf16(false, a, false, b, (short)0, c,
                                                   false, false);
}

// ---------------- weight convert + transpose: W[K][N] f32 -> Wt[N][K] bf16 ----
__global__ void cvt_tr_k(const float* __restrict__ W, bf16* __restrict__ Wt,
                         int K, int Nc) {
    int idx = blockIdx.x * 256 + threadIdx.x;
    if (idx >= K * Nc) return;
    int k = idx / Nc, n = idx % Nc;
    Wt[(size_t)n * K + k] = (bf16)W[idx];
}

// ---------------- LayerNorm: f32 [rows, 512] -> bf16 [rows, 512] --------------
__global__ __launch_bounds__(256) void ln_k(const float* __restrict__ in,
                                            const float* __restrict__ g,
                                            const float* __restrict__ be,
                                            bf16* __restrict__ out) {
    __shared__ float red[16];
    const int row = blockIdx.x, t = threadIdx.x;
    const float* p = in + (size_t)row * DD;
    float v0 = p[t], v1 = p[t + 256];
    float s = v0 + v1, ss = v0 * v0 + v1 * v1;
#pragma unroll
    for (int m = 16; m >= 1; m >>= 1) {
        s  += __shfl_xor(s, m);
        ss += __shfl_xor(ss, m);
    }
    const int wave = t >> 5, lane = t & 31;
    if (lane == 0) { red[wave] = s; red[8 + wave] = ss; }
    __syncthreads();
    float ts = 0.f, tss = 0.f;
#pragma unroll
    for (int i = 0; i < 8; ++i) { ts += red[i]; tss += red[8 + i]; }
    const float mu   = ts * (1.0f / DD);
    const float var  = tss * (1.0f / DD) - mu * mu;
    const float rstd = rsqrtf(var + 1e-6f);
    bf16* o = out + (size_t)row * DD;
    o[t]       = (bf16)((v0 - mu) * rstd * g[t]       + be[t]);
    o[t + 256] = (bf16)((v1 - mu) * rstd * g[t + 256] + be[t + 256]);
}

// ---------------- bf16 WMMA GEMM, wave = 64x16 output tile --------------------
// MODE 0: out bf16 = (A@Wt + bias) * mask[row]                  (Q, K proj)
// MODE 1: like 0 but scattered into vT layout [B, Nc, rowsPerB] (V proj)
// MODE 2: out f32  = A@Wt + bias + resid                        (Wo, FFN2)
// MODE 3: out bf16 = gelu(A@Wt + bias)                          (FFN1)
template <int MODE>
__global__ __launch_bounds__(256) void gemm_bf16_k(
    const bf16* __restrict__ A, const bf16* __restrict__ Bt,
    const float* __restrict__ bias, const float* __restrict__ mask,
    const float* __restrict__ resid, void* __restrict__ outp,
    int Mrows, int Nc, int K, int rowsPerB) {
    const int lane = threadIdx.x & 31;
    const int wave = threadIdx.x >> 5;
    const int r  = lane & 15;
    const int kb = (lane >> 4) << 3;
    const int hl = lane >> 4;
    const int n0 = (blockIdx.x * 8 + wave) * 16;
    const int m0 = blockIdx.y * 64;
    if (n0 >= Nc || m0 >= Mrows) return;

    v8f acc0 = {}, acc1 = {}, acc2 = {}, acc3 = {};
    const bf16* bptr = Bt + (size_t)(n0 + r) * K + kb;
    const bf16* aptr = A  + (size_t)(m0 + r) * K + kb;
    const size_t a16 = (size_t)16 * K;

    for (int kk = 0; kk < K; kk += 32) {
        v16bf bf = frag_ld(bptr + kk);
        v16bf a0 = frag_ld(aptr + kk);
        v16bf a1 = frag_ld(aptr + a16 + kk);
        v16bf a2 = frag_ld(aptr + 2 * a16 + kk);
        v16bf a3 = frag_ld(aptr + 3 * a16 + kk);
        acc0 = wmma_bf16(a0, bf, acc0);
        acc1 = wmma_bf16(a1, bf, acc1);
        acc2 = wmma_bf16(a2, bf, acc2);
        acc3 = wmma_bf16(a3, bf, acc3);
    }

    const int col = n0 + r;
    const float bcol = bias[col];
#pragma unroll
    for (int i = 0; i < 4; ++i) {
        v8f acc = (i == 0) ? acc0 : (i == 1) ? acc1 : (i == 2) ? acc2 : acc3;
#pragma unroll
        for (int j = 0; j < 8; ++j) {
            const int row = m0 + i * 16 + j + (hl << 3);
            float val = acc[j] + bcol;
            if constexpr (MODE == 0) {
                val *= mask[row];
                ((bf16*)outp)[(size_t)row * Nc + col] = (bf16)val;
            } else if constexpr (MODE == 1) {
                val *= mask[row];
                const int b = row / rowsPerB, mloc = row % rowsPerB;
                ((bf16*)outp)[((size_t)b * Nc + col) * rowsPerB + mloc] = (bf16)val;
            } else if constexpr (MODE == 2) {
                ((float*)outp)[(size_t)row * Nc + col] =
                    val + resid[(size_t)row * Nc + col];
            } else {  // MODE 3: exact GELU
                const float gv = 0.5f * val * (1.0f + erff(val * 0.70710678118f));
                ((bf16*)outp)[(size_t)row * Nc + col] = (bf16)gv;
            }
        }
    }
}

// ---------------- flash attention: wave = (b, h, 16-row q tile) ---------------
__global__ __launch_bounds__(256) void attn_k(const bf16* __restrict__ q,
                                              const bf16* __restrict__ kmat,
                                              const bf16* __restrict__ vT,
                                              bf16* __restrict__ ctx) {
    __shared__ bf16 pst[8 * 16 * 32];  // per-wave 16x32 P staging
    const int lane = threadIdx.x & 31, wave = threadIdx.x >> 5;
    const int r  = lane & 15;
    const int kb = (lane >> 4) << 3;
    const int hl = lane >> 4;
    const int tile = blockIdx.x * 8 + wave;
    const int QT = NN / 16;                   // 128
    const int qt = tile % QT;
    const int h  = (tile / QT) % HH;
    const int b  = tile / (QT * HH);

    const bf16* qbase = q + ((size_t)(b * NN + qt * 16 + r)) * DD + h * DH + kb;
    const v16bf qf0 = frag_ld(qbase);
    const v16bf qf1 = frag_ld(qbase + 32);

    float mrow[8], lrow[8];
#pragma unroll
    for (int j = 0; j < 8; ++j) { mrow[j] = -1e30f; lrow[j] = 0.f; }
    v8f c0 = {}, c1 = {}, c2 = {}, c3 = {};
    bf16* pw = pst + wave * 512;

    for (int kc = 0; kc < MM; kc += 32) {
        v8f s0 = {}, s1 = {};
        {
            const bf16* k0 = kmat + ((size_t)(b * MM + kc + r)) * DD + h * DH + kb;
            const bf16* k1 = k0 + (size_t)16 * DD;
            s0 = wmma_bf16(qf0, frag_ld(k0), s0);
            s0 = wmma_bf16(qf1, frag_ld(k0 + 32), s0);
            s1 = wmma_bf16(qf0, frag_ld(k1), s1);
            s1 = wmma_bf16(qf1, frag_ld(k1 + 32), s1);
        }
        float csc[8];
#pragma unroll
        for (int j = 0; j < 8; ++j) {
            const float a = s0[j] * 0.125f;     // 1/sqrt(64)
            const float e = s1[j] * 0.125f;
            float rm = fmaxf(a, e);
            rm = fmaxf(rm, __shfl_xor(rm, 1));
            rm = fmaxf(rm, __shfl_xor(rm, 2));
            rm = fmaxf(rm, __shfl_xor(rm, 4));
            rm = fmaxf(rm, __shfl_xor(rm, 8));
            const float nm = fmaxf(mrow[j], rm);
            const float sc = __expf(mrow[j] - nm);
            const float p0 = __expf(a - nm);
            const float p1 = __expf(e - nm);
            float rs = p0 + p1;
            rs += __shfl_xor(rs, 1);
            rs += __shfl_xor(rs, 2);
            rs += __shfl_xor(rs, 4);
            rs += __shfl_xor(rs, 8);
            lrow[j] = lrow[j] * sc + rs;
            mrow[j] = nm;
            csc[j]  = sc;
            const int row = j + (hl << 3);
            pw[row * 32 + r]      = (bf16)p0;
            pw[row * 32 + 16 + r] = (bf16)p1;
        }
#pragma unroll
        for (int j = 0; j < 8; ++j) {
            c0[j] *= csc[j]; c1[j] *= csc[j];
            c2[j] *= csc[j]; c3[j] *= csc[j];
        }
        __builtin_amdgcn_wave_barrier();
        const v16bf pa = frag_ld(pw + r * 32 + kb);
        __builtin_amdgcn_wave_barrier();
        const bf16* vb = vT + ((size_t)(b * DD + h * DH + r)) * MM + kc + kb;
        c0 = wmma_bf16(pa, frag_ld(vb), c0);
        c1 = wmma_bf16(pa, frag_ld(vb + (size_t)16 * MM), c1);
        c2 = wmma_bf16(pa, frag_ld(vb + (size_t)32 * MM), c2);
        c3 = wmma_bf16(pa, frag_ld(vb + (size_t)48 * MM), c3);
    }
#pragma unroll
    for (int j = 0; j < 8; ++j) {
        const float inv = 1.0f / lrow[j];
        const int row = b * NN + qt * 16 + j + (hl << 3);
        bf16* cp = ctx + (size_t)row * DD + h * DH + r;
        cp[0]  = (bf16)(c0[j] * inv);
        cp[16] = (bf16)(c1[j] * inv);
        cp[32] = (bf16)(c2[j] * inv);
        cp[48] = (bf16)(c3[j] * inv);
    }
}

// ---------------- orchestration ----------------------------------------------
extern "C" void kernel_launch(void* const* d_in, const int* in_sizes, int n_in,
                              void* d_out, int out_size, void* d_ws, size_t ws_size,
                              hipStream_t stream) {
    const float* x      = (const float*)d_in[0];
    const float* y      = (const float*)d_in[1];
    const float* mask_x = (const float*)d_in[2];
    const float* mask_y = (const float*)d_in[3];
    const float* Wq = (const float*)d_in[4];  const float* bq = (const float*)d_in[5];
    const float* Wk = (const float*)d_in[6];  const float* bk = (const float*)d_in[7];
    const float* Wv = (const float*)d_in[8];  const float* bv = (const float*)d_in[9];
    const float* Wo = (const float*)d_in[10]; const float* bo = (const float*)d_in[11];
    const float* W1 = (const float*)d_in[12]; const float* b1 = (const float*)d_in[13];
    const float* W2 = (const float*)d_in[14]; const float* b2 = (const float*)d_in[15];
    const float* ln1_g = (const float*)d_in[16]; const float* ln1_b = (const float*)d_in[17];
    const float* ln2_g = (const float*)d_in[18]; const float* ln2_b = (const float*)d_in[19];

    char* p = (char*)d_ws;
    auto alloc = [&](size_t bytes) -> char* {
        char* ret = p;
        p += (bytes + 255) & ~(size_t)255;
        return ret;
    };
    bf16* wqT = (bf16*)alloc((size_t)DD * DD * 2);
    bf16* wkT = (bf16*)alloc((size_t)DD * DD * 2);
    bf16* wvT = (bf16*)alloc((size_t)DD * DD * 2);
    bf16* woT = (bf16*)alloc((size_t)DD * DD * 2);
    bf16* w1T = (bf16*)alloc((size_t)DD * FF * 2);
    bf16* w2T = (bf16*)alloc((size_t)FF * DD * 2);
    bf16* xn  = (bf16*)alloc((size_t)ROWS * DD * 2);
    bf16* yn  = (bf16*)alloc((size_t)ROWS * DD * 2);
    bf16* qb  = (bf16*)alloc((size_t)ROWS * DD * 2);
    bf16* kbuf= (bf16*)alloc((size_t)ROWS * DD * 2);
    bf16* vTb = (bf16*)alloc((size_t)ROWS * DD * 2);
    bf16* ctx = (bf16*)alloc((size_t)ROWS * DD * 2);
    float* x1 = (float*)alloc((size_t)ROWS * DD * 4);
    bf16* xn2 = (bf16*)alloc((size_t)ROWS * DD * 2);
    // ffh reuses the (dead-by-then) xn/yn/qb/kbuf region: 4*4MB = 16MB needed.
    bf16* ffh = xn;

    // 1) weight convert+transpose (f32 [K,N] -> bf16 [N,K])
    {
        int t = DD * DD;
        cvt_tr_k<<<(t + 255) / 256, 256, 0, stream>>>(Wq, wqT, DD, DD);
        cvt_tr_k<<<(t + 255) / 256, 256, 0, stream>>>(Wk, wkT, DD, DD);
        cvt_tr_k<<<(t + 255) / 256, 256, 0, stream>>>(Wv, wvT, DD, DD);
        cvt_tr_k<<<(t + 255) / 256, 256, 0, stream>>>(Wo, woT, DD, DD);
        int t2 = DD * FF;
        cvt_tr_k<<<(t2 + 255) / 256, 256, 0, stream>>>(W1, w1T, DD, FF);
        cvt_tr_k<<<(t2 + 255) / 256, 256, 0, stream>>>(W2, w2T, FF, DD);
    }
    // 2) shared attention LayerNorm on x and y
    ln_k<<<ROWS, 256, 0, stream>>>(x, ln1_g, ln1_b, xn);
    ln_k<<<ROWS, 256, 0, stream>>>(y, ln1_g, ln1_b, yn);

    // 3) masked Q/K/V projections
    dim3 gD(DD / 128, ROWS / 64);   // (4, 64)
    gemm_bf16_k<0><<<gD, 256, 0, stream>>>(xn, wqT, bq, mask_x, nullptr, qb,
                                           ROWS, DD, DD, 0);
    gemm_bf16_k<0><<<gD, 256, 0, stream>>>(yn, wkT, bk, mask_y, nullptr, kbuf,
                                           ROWS, DD, DD, 0);
    gemm_bf16_k<1><<<gD, 256, 0, stream>>>(yn, wvT, bv, mask_y, nullptr, vTb,
                                           ROWS, DD, DD, MM);

    // 4) flash attention (one wave per (b,h,q16))
    attn_k<<<(BB * HH * (NN / 16)) / 8, 256, 0, stream>>>(qb, kbuf, vTb, ctx);

    // 5) output projection + residual -> x1 (f32)
    gemm_bf16_k<2><<<gD, 256, 0, stream>>>(ctx, woT, bo, nullptr, x, x1,
                                           ROWS, DD, DD, 0);

    // 6) ffn_norm + MLP with residual -> d_out (f32)
    ln_k<<<ROWS, 256, 0, stream>>>(x1, ln2_g, ln2_b, xn2);
    dim3 gF(FF / 128, ROWS / 64);   // (16, 64)
    gemm_bf16_k<3><<<gF, 256, 0, stream>>>(xn2, w1T, b1, nullptr, nullptr, ffh,
                                           ROWS, FF, DD, 0);
    gemm_bf16_k<2><<<gD, 256, 0, stream>>>(ffh, w2T, b2, nullptr, x1, d_out,
                                           ROWS, DD, FF, 0);
}
